// HTGNN_NoTemporal_3006477107342
// MI455X (gfx1250) — matrix operand order repaired and checked by reading, not compiled, not run on hardware
//
#include <hip/hip_runtime.h>
#include <hip/hip_bf16.h>
#include <math.h>

// ---------------- constants from the reference ----------------
#define NN     50000
#define DD     128
#define HH     8
#define DHH    16
#define RR     3
#define EE     400000
#define NEG_SLOPE 0.2f
#define LN_EPS    1e-5f

typedef _Float16 h8_t  __attribute__((ext_vector_type(8)));
typedef _Float16 v16h  __attribute__((ext_vector_type(16)));
typedef float    v8f   __attribute__((ext_vector_type(8)));
typedef unsigned int u32x4 __attribute__((ext_vector_type(4)));
typedef int      i32x4 __attribute__((ext_vector_type(4)));
typedef int      i32x8 __attribute__((ext_vector_type(8)));

#define USE_TDM 1

// ---------------- small helpers ----------------
__device__ __forceinline__ void atomicMaxFloat(float* addr, float val) {
    if (val >= 0.0f) {
        atomicMax((int*)addr, __float_as_int(val));
    } else {
        atomicMin((unsigned int*)addr, __float_as_uint(val));
    }
}

// f32 -> f16 conversion
__global__ void k_f32_to_f16(const float* __restrict__ src, _Float16* __restrict__ dst, int n) {
    int i = blockIdx.x * blockDim.x + threadIdx.x;
    if (i < n) dst[i] = (_Float16)src[i];
}

// W[k][n] (128x128 f32) -> Wt[n][k] f16
__global__ void k_transpose_w(const float* __restrict__ W, _Float16* __restrict__ Wt) {
    int i = blockIdx.x * blockDim.x + threadIdx.x;  // i < 128*128
    if (i >= DD * DD) return;
    int n = i >> 7, k = i & 127;
    Wt[n * DD + k] = (_Float16)W[k * DD + n];
}

__global__ void k_fill(float* __restrict__ p, float v, int n) {
    int i = blockIdx.x * blockDim.x + threadIdx.x;
    if (i < n) p[i] = v;
}

// ---------------- WMMA GEMM: C[M,128] = A[M,128] (f16) x B[128,128] ----------------
// Bt stored transposed: Bt[n][k] f16.  Bt is staged into LDS once per block via
// the Tensor Data Mover, then each wave computes 16 rows x 128 cols per M-tile.
__global__ __launch_bounds__(256)
void k_gemm_wmma(const _Float16* __restrict__ A, const _Float16* __restrict__ Bt,
                 float* __restrict__ C, int mtiles) {
    __shared__ _Float16 Bs[DD * DD];          // 32 KB weight tile

#if USE_TDM
    // ---- Tensor DMA: load Bt (128x128 f16, 32KB) global -> LDS (wave 0 issues) ----
    if ((threadIdx.x >> 5) == 0) {
        unsigned long long ga = (unsigned long long)(const void*)Bt;
        unsigned int lds = (unsigned int)(size_t)(void*)&Bs[0];  // LDS byte offset
        u32x4 g0;
        g0[0] = 1u;                                   // count=1, no gather
        g0[1] = lds;                                  // lds_addr [63:32]
        g0[2] = (unsigned int)(ga & 0xFFFFFFFFu);     // global_addr lo
        g0[3] = (unsigned int)((ga >> 32) & 0x1FFFFFFu) | (2u << 30); // addr hi | type=2
        i32x8 g1;
        g1[0] = 0x10000;        // workgroup_mask=0, data_size=1 (2 bytes)
        g1[1] = 128 << 16;      // tensor_dim0[15:0]=128 (bits 48..63)
        g1[2] = 128 << 16;      // tensor_dim1[15:0]=128 (bits 80..95)
        g1[3] = 128 << 16;      // tile_dim0=128 (bits 112..127)
        g1[4] = 128;            // tile_dim1=128 (bits 128..143)
        g1[5] = 128;            // tensor_dim0_stride=128 (bits 160..191)
        g1[6] = 0;
        g1[7] = 0;
        i32x4 gz4 = {0, 0, 0, 0};
        i32x8 gz8 = {0, 0, 0, 0, 0, 0, 0, 0};
        __builtin_amdgcn_tensor_load_to_lds(g0, g1, gz4, gz4, gz8, 0);
        __builtin_amdgcn_s_wait_tensorcnt(0);
    }
    __syncthreads();
#else
    for (int i = threadIdx.x; i < DD * DD / 8; i += blockDim.x) {
        reinterpret_cast<h8_t*>(Bs)[i] = reinterpret_cast<const h8_t*>(Bt)[i];
    }
    __syncthreads();
#endif

    const int lane   = threadIdx.x & 31;
    const int l15    = lane & 15;
    const int hiHalf = (lane >> 4) & 1;       // 0: lanes 0-15, 1: lanes 16-31
    const int kbaseA = hiHalf ? 16 : 0;       // 16-bit A fragment layout (contiguous k)
    const int kbaseB = hiHalf ? 16 : 0;       // 16-bit B fragment layout
    const int nwaves = gridDim.x * (blockDim.x >> 5);

    for (int wave = blockIdx.x * (blockDim.x >> 5) + (threadIdx.x >> 5);
         wave < mtiles; wave += nwaves) {
        const size_t rowBase = (size_t)(wave * 16 + l15) * DD;

        v8f acc[8];
#pragma unroll
        for (int i = 0; i < 8; ++i) acc[i] = (v8f)(0.0f);

#pragma unroll
        for (int kt = 0; kt < 4; ++kt) {
            const int k0 = kt * 32;
            union { v16h v; struct { h8_t lo, hi; } p; } afr;
            afr.p.lo = *reinterpret_cast<const h8_t*>(A + rowBase + k0 + kbaseA);
            afr.p.hi = *reinterpret_cast<const h8_t*>(A + rowBase + k0 + 8 + kbaseA);
#pragma unroll
            for (int nt = 0; nt < 8; ++nt) {
                const int n = nt * 16 + l15;
                v16h bfr = *reinterpret_cast<const v16h*>(Bs + n * DD + k0 + kbaseB);
                acc[nt] = __builtin_amdgcn_wmma_f32_16x16x32_f16(
                    false, afr.v, false, bfr, (short)0, acc[nt], false, false);
            }
        }

        // C/D layout: lane l15 -> column in tile; VGPR v -> row v (+8 for hi lanes)
        const int mrow0 = wave * 16 + (hiHalf ? 8 : 0);
#pragma unroll
        for (int nt = 0; nt < 8; ++nt) {
            const int n = nt * 16 + l15;
#pragma unroll
            for (int v = 0; v < 8; ++v) {
                C[(size_t)(mrow0 + v) * DD + n] = acc[nt][v];
            }
        }
    }
}

// ---------------- attention scalar terms: el/er [N,H] ----------------
__global__ void k_attn_terms(const float* __restrict__ hbuf,
                             const float* __restrict__ al, const float* __restrict__ ar,
                             float* __restrict__ el, float* __restrict__ er) {
    int t = blockIdx.x * blockDim.x + threadIdx.x;   // t < N*H
    if (t >= NN * HH) return;
    int node = t >> 3, head = t & 7;
    const float* hp  = hbuf + (size_t)node * DD + head * DHH;
    const float* alp = al + head * DHH;
    const float* arp = ar + head * DHH;
    float sl = 0.f, sr = 0.f;
#pragma unroll
    for (int d = 0; d < DHH; ++d) { sl += hp[d] * alp[d]; sr += hp[d] * arp[d]; }
    el[t] = sl; er[t] = sr;
}

// ---------------- edge pass 1: segment max of leaky-relu scores ----------------
__global__ void k_edge_max(const float* __restrict__ el, const float* __restrict__ er,
                           const int* __restrict__ src, const int* __restrict__ dst,
                           float* __restrict__ mmax) {
    int e = blockIdx.x * blockDim.x + threadIdx.x;
    if (e >= EE) return;
    int s = src[e], d = dst[e];
#pragma unroll
    for (int h = 0; h < HH; ++h) {
        float v = el[s * HH + h] + er[d * HH + h];
        v = v > 0.f ? v : v * NEG_SLOPE;
        atomicMaxFloat(&mmax[d * HH + h], v);
    }
}

// ---------------- edge pass 2: exp(e - max) and denominator ----------------
__global__ void k_edge_exp(const float* __restrict__ el, const float* __restrict__ er,
                           const int* __restrict__ src, const int* __restrict__ dst,
                           const float* __restrict__ mmax, float* __restrict__ denom,
                           float* __restrict__ eattn) {
    int e = blockIdx.x * blockDim.x + threadIdx.x;
    if (e >= EE) return;
    int s = src[e], d = dst[e];
#pragma unroll
    for (int h = 0; h < HH; ++h) {
        float v = el[s * HH + h] + er[d * HH + h];
        v = v > 0.f ? v : v * NEG_SLOPE;
        float ex = __expf(v - mmax[d * HH + h]);
        eattn[(size_t)e * HH + h] = ex;
        atomicAdd(&denom[d * HH + h], ex);
    }
}

// ---------------- edge pass 3: weighted message aggregation ----------------
// one 128-thread block per edge: lane c = feature channel (coalesced atomics)
__global__ __launch_bounds__(128)
void k_edge_msg(const float* __restrict__ eattn, const float* __restrict__ denom,
                const int* __restrict__ src, const int* __restrict__ dst,
                const float* __restrict__ hbuf, float* __restrict__ acc) {
    int e = blockIdx.x;
    int c = threadIdx.x;
    int s = src[e], d = dst[e];
    int head = c >> 4;
    float alpha = eattn[(size_t)e * HH + head] / denom[d * HH + head];
    atomicAdd(&acc[(size_t)d * DD + c], alpha * hbuf[(size_t)s * DD + c]);
}

// ---------------- layer epilogues ----------------
__global__ void k_finalize1(const float* __restrict__ acc, const float* __restrict__ b,
                            float* __restrict__ h1) {
    int i = blockIdx.x * blockDim.x + threadIdx.x;
    if (i >= NN * DD) return;
    int c = i & 127;
    float bs = b[c] + b[DD + c] + b[2 * DD + c];
    float v = acc[i] + bs;
    h1[i] = v > 0.f ? v : 0.f;
}

__global__ void k_finalize2(const float* __restrict__ acc, const float* __restrict__ b,
                            const float* __restrict__ h1, float* __restrict__ hres) {
    int i = blockIdx.x * blockDim.x + threadIdx.x;
    if (i >= NN * DD) return;
    int c = i & 127;
    float bs = b[c] + b[DD + c] + b[2 * DD + c];
    hres[i] = acc[i] + bs + h1[i];
}

// ---------------- LayerNorm (block 128 = one node) ----------------
__global__ __launch_bounds__(128)
void k_layernorm(const float* __restrict__ h, const float* __restrict__ g,
                 const float* __restrict__ b, float* __restrict__ out) {
    __shared__ float red[128];
    int n = blockIdx.x, c = threadIdx.x;
    float v = h[(size_t)n * DD + c];
    red[c] = v; __syncthreads();
    for (int s = 64; s > 0; s >>= 1) { if (c < s) red[c] += red[c + s]; __syncthreads(); }
    float mu = red[0] * (1.0f / DD);
    __syncthreads();
    float dv = v - mu;
    red[c] = dv * dv; __syncthreads();
    for (int s = 64; s > 0; s >>= 1) { if (c < s) red[c] += red[c + s]; __syncthreads(); }
    float var = red[0] * (1.0f / DD);
    out[(size_t)n * DD + c] = dv * rsqrtf(var + LN_EPS) * g[c] + b[c];
}

// ---------------- classifier ----------------
__global__ void k_bias_relu(float* __restrict__ x, const float* __restrict__ b) {
    int i = blockIdx.x * blockDim.x + threadIdx.x;
    if (i >= NN * DD) return;
    float v = x[i] + b[i & 127];
    x[i] = v > 0.f ? v : 0.f;
}

__global__ void k_final_dot(const float* __restrict__ hc, const float* __restrict__ w,
                            const float* __restrict__ b, float* __restrict__ out) {
    int n = blockIdx.x * blockDim.x + threadIdx.x;
    if (n >= NN) return;
    const float* hp = hc + (size_t)n * DD;
    float s = 0.f;
#pragma unroll 4
    for (int d = 0; d < DD; ++d) s += hp[d] * w[d];
    out[n] = s + b[0];
}

// ---------------- orchestration ----------------
extern "C" void kernel_launch(void* const* d_in, const int* in_sizes, int n_in,
                              void* d_out, int out_size, void* d_ws, size_t ws_size,
                              hipStream_t stream) {
    (void)in_sizes; (void)n_in; (void)out_size; (void)ws_size;

    const float* feat = (const float*)d_in[0];
    const int*   esrc = (const int*)d_in[1];
    const int*   edst = (const int*)d_in[2];
    const float* W1  = (const float*)d_in[3];
    const float* al1 = (const float*)d_in[4];
    const float* ar1 = (const float*)d_in[5];
    const float* b1  = (const float*)d_in[6];
    const float* W2  = (const float*)d_in[7];
    const float* al2 = (const float*)d_in[8];
    const float* ar2 = (const float*)d_in[9];
    const float* b2  = (const float*)d_in[10];
    const float* lng = (const float*)d_in[11];
    const float* lnb = (const float*)d_in[12];
    const float* Wc1 = (const float*)d_in[13];
    const float* bc1 = (const float*)d_in[14];
    const float* Wc2 = (const float*)d_in[15];
    const float* bc2 = (const float*)d_in[16];
    float* out = (float*)d_out;

    // workspace carve-up (256B aligned pieces)
    char* ws = (char*)d_ws;
    size_t off = 0;
    auto carve = [&](size_t bytes) { char* p = ws + off; off += (bytes + 255) & ~(size_t)255; return p; };
    _Float16* xh    = (_Float16*)carve((size_t)NN * DD * 2);   // f16 input copy
    _Float16* wt    = (_Float16*)carve((size_t)DD * DD * 2);   // f16 transposed weight
    float*    hbuf  = (float*)carve((size_t)NN * DD * 4);      // per-relation projected feats
    float*    el    = (float*)carve((size_t)NN * HH * 4);
    float*    er    = (float*)carve((size_t)NN * HH * 4);
    float*    mmax  = (float*)carve((size_t)NN * HH * 4);
    float*    denom = (float*)carve((size_t)NN * HH * 4);
    float*    eattn = (float*)carve((size_t)EE * HH * 4);
    float*    acc   = (float*)carve((size_t)NN * DD * 4);
    float*    h1    = (float*)carve((size_t)NN * DD * 4);
    float*    hres  = (float*)carve((size_t)NN * DD * 4);

    const int mtiles = NN / 16;                    // 3125 M-tiles
    const dim3 gGemm((mtiles + 7) / 8), bGemm(256);  // 8 waves/block, persistent
    const dim3 gND((NN * DD + 255) / 256), b256(256);
    const dim3 gNH((NN * HH + 255) / 256);
    const dim3 gE((EE + 255) / 256);
    const dim3 gTW((DD * DD + 255) / 256);

    auto run_layer = [&](const float* x, const float* W, const float* al,
                         const float* ar) {
        (void)hipMemsetAsync(acc, 0, (size_t)NN * DD * 4, stream);
        k_f32_to_f16<<<gND, b256, 0, stream>>>(x, xh, NN * DD);
        for (int r = 0; r < RR; ++r) {
            const int* sr = esrc + (size_t)r * EE;
            const int* dr = edst + (size_t)r * EE;
            k_transpose_w<<<gTW, b256, 0, stream>>>(W + (size_t)r * DD * DD, wt);
            k_gemm_wmma<<<gGemm, bGemm, 0, stream>>>(xh, wt, hbuf, mtiles);
            k_attn_terms<<<gNH, b256, 0, stream>>>(hbuf, al + r * HH * DHH,
                                                   ar + r * HH * DHH, el, er);
            k_fill<<<(NN * HH + 255) / 256, 256, 0, stream>>>(mmax, -INFINITY, NN * HH);
            (void)hipMemsetAsync(denom, 0, (size_t)NN * HH * 4, stream);
            k_edge_max<<<gE, b256, 0, stream>>>(el, er, sr, dr, mmax);
            k_edge_exp<<<gE, b256, 0, stream>>>(el, er, sr, dr, mmax, denom, eattn);
            k_edge_msg<<<EE, 128, 0, stream>>>(eattn, denom, sr, dr, hbuf, acc);
        }
    };

    // layer 1: h1 = relu(sum_r gat_r(feat) + sum_r b1_r)
    run_layer(feat, W1, al1, ar1);
    k_finalize1<<<gND, b256, 0, stream>>>(acc, b1, h1);

    // layer 2: hres = sum_r gat_r(h1) + sum_r b2_r + h1
    run_layer(h1, W2, al2, ar2);
    k_finalize2<<<gND, b256, 0, stream>>>(acc, b2, h1, hres);

    // LayerNorm -> acc (reused as hln)
    k_layernorm<<<NN, 128, 0, stream>>>(hres, lng, lnb, acc);

    // classifier: hc = relu(hln @ Wc1 + bc1); out = hc @ Wc2 + bc2
    k_f32_to_f16<<<gND, b256, 0, stream>>>(acc, xh, NN * DD);
    k_transpose_w<<<gTW, b256, 0, stream>>>(Wc1, wt);
    k_gemm_wmma<<<gGemm, bGemm, 0, stream>>>(xh, wt, hbuf, mtiles);
    k_bias_relu<<<gND, b256, 0, stream>>>(hbuf, bc1);
    k_final_dot<<<(NN + 255) / 256, 256, 0, stream>>>(hbuf, Wc2, bc2, out);
}
